// VanillaLSTM_33337536152171
// MI455X (gfx1250) — matrix operand order repaired and checked by reading, not compiled, not run on hardware
//
#include <hip/hip_runtime.h>
#include <hip/hip_bf16.h>

// ---------------------------------------------------------------------------
// VanillaLSTM forward for MI455X (gfx1250), wave32 + WMMA bf16 + async LDS.
//   B=64, S=512, H=1024, G=4H=4096, A=128 actions.
// One WMMA fragment = 16x16(x32K) tile = 32 lanes x 16 bf16 = 1024 bytes.
// ---------------------------------------------------------------------------

#define LSTM_B   64
#define LSTM_S   512
#define LSTM_H   1024
#define LSTM_G   4096
#define LSTM_A   128
#define FRAG_USH 512                 // ushorts per fragment (32 lanes * 16)
#define SLOT_USH (4 * 32 * FRAG_USH) // one h slot: 4 m-tiles x 32 k-tiles
#define KT_CHUNK 16                  // k-tiles staged per LDS chunk

typedef __attribute__((ext_vector_type(16))) __bf16        v16bf;
typedef __attribute__((ext_vector_type(8)))  float         v8f;
typedef __attribute__((ext_vector_type(8)))  unsigned int  v8u;

static __device__ __forceinline__ unsigned short f32_to_bf16_rne(float f) {
  unsigned int u = __float_as_uint(f);
  unsigned int r = u + 0x7fffu + ((u >> 16) & 1u);
  return (unsigned short)(r >> 16);
}

static __device__ __forceinline__ v16bf load_frag(const unsigned short* p) {
  return __builtin_bit_cast(v16bf, *(const v8u*)p);
}

// A-operand (16-bit, 16x32 MxK) per-lane element j -> k_local map (ISA 7.12.2)
static __device__ __forceinline__ int a_kmap(int lane, int j) {
  return j + 8 * ((j >> 3) + (lane >> 4));
}
// B-operand (16-bit, 32x16 KxN): same k map as A (lanes 0-15 K=0..7,16..23; lanes 16-31 K=8..15,24..31)
static __device__ __forceinline__ int b_kmap(int lane, int j) {
  return j + 8 * ((j >> 3) + (lane >> 4));
}

// Async copy of 16 bytes global -> LDS (GLOBAL_LOAD_ASYNC_TO_LDS_B128, GV mode).
static __device__ __forceinline__ void async_g2l_b128(unsigned int lds_byte_off,
                                                      const void* gptr) {
  unsigned long long ga = (unsigned long long)gptr;
  asm volatile("global_load_async_to_lds_b128 %0, %1, off"
               :: "v"(lds_byte_off), "v"(ga) : "memory");
}

static __device__ __forceinline__ void wait_async0() {
  asm volatile("s_wait_asynccnt 0" ::: "memory");
}

// ---------------------------------------------------------------------------
__global__ void zero_u32_kernel(unsigned int* __restrict__ p, long n) {
  long i = (long)blockIdx.x * blockDim.x + threadIdx.x;
  if (i < n) p[i] = 0u;
}

// Gather embedding rows, convert f32->bf16, pack as WMMA A fragments.
// Row r = s*64 + b  (so step t owns row-tiles t*4 .. t*4+3).
__global__ void gather_pack_A_kernel(const float* __restrict__ embed,
                                     const int* __restrict__ x,
                                     unsigned short* __restrict__ Ap) {
  long tid  = (long)blockIdx.x * blockDim.x + threadIdx.x; // 2048*32*32 threads
  int  lane = (int)(tid & 31);
  long frag = tid >> 5;
  int  kt   = (int)(frag & 31);
  long rt   = frag >> 5;
  int  m    = lane & 15;
  long r    = rt * 16 + m;          // = s*64 + b
  int  s    = (int)(r >> 6);
  int  b    = (int)(r & 63);
  int  token = x[b * LSTM_S + s];
  const float* erow = embed + (long)token * LSTM_H;
  unsigned short vals[16];
#pragma unroll
  for (int j = 0; j < 16; ++j) {
    int k = kt * 32 + a_kmap(lane, j);
    vals[j] = f32_to_bf16_rne(erow[k]);
  }
  *(v8u*)(Ap + tid * 16) = *(const v8u*)vals;
}

// Pack [Wi ; Wh] (K=2048 x N=4096) as WMMA B fragments.
// Frag index = nt*64 + kt  (nt in 0..255, kt in 0..63; kt<32 -> Wi, else Wh).
__global__ void pack_Wboth_kernel(const float* __restrict__ Wi,
                                  const float* __restrict__ Wh,
                                  unsigned short* __restrict__ Wp) {
  long tid  = (long)blockIdx.x * blockDim.x + threadIdx.x; // 256*64*32 threads
  int  lane = (int)(tid & 31);
  long frag = tid >> 5;
  int  kt   = (int)(frag & 63);
  int  nt   = (int)(frag >> 6);
  int  n    = nt * 16 + (lane & 15);
  unsigned short vals[16];
#pragma unroll
  for (int j = 0; j < 16; ++j) {
    int k = kt * 32 + b_kmap(lane, j);
    float w = (k < LSTM_H) ? Wi[(long)k * LSTM_G + n]
                           : Wh[(long)(k - LSTM_H) * LSTM_G + n];
    vals[j] = f32_to_bf16_rne(w);
  }
  *(v8u*)(Wp + tid * 16) = *(const v8u*)vals;
}

// Pack Wo (K=1024 x N=128) as WMMA B fragments. Frag index = nt*32 + kt.
__global__ void pack_Wo_kernel(const float* __restrict__ Wo,
                               unsigned short* __restrict__ Wop) {
  long tid  = (long)blockIdx.x * blockDim.x + threadIdx.x; // 8*32*32 threads
  int  lane = (int)(tid & 31);
  long frag = tid >> 5;
  int  kt   = (int)(frag & 31);
  int  nt   = (int)(frag >> 5);
  int  n    = nt * 16 + (lane & 15);
  unsigned short vals[16];
#pragma unroll
  for (int j = 0; j < 16; ++j) {
    int k = kt * 32 + b_kmap(lane, j);
    vals[j] = f32_to_bf16_rne(Wo[(long)k * LSTM_A + n]);
  }
  *(v8u*)(Wop + tid * 16) = *(const v8u*)vals;
}

// One LSTM step's gate GEMM:  gates[64,4096] = [x_emb_t ; h_{t-1}] @ [Wi;Wh] + bh
// Block = 4 waves (one per m-tile) owning an n-pair; B fragments staged in LDS
// via async global->LDS copies, double-buffered against the WMMA K-loop.
__global__ __launch_bounds__(128)
void lstm_gates_gemm_kernel(const unsigned short* __restrict__ Ap,
                            const unsigned short* __restrict__ Wp,
                            const unsigned short* __restrict__ hprev,
                            const float* __restrict__ bh,
                            float* __restrict__ gates, int t) {
  // [buf][ntp][ktl][lane*16] -> 2 * 2*16 frags * 1KB = 64 KB
  __shared__ unsigned short Bs[2][2 * KT_CHUNK * FRAG_USH];
  int tid  = threadIdx.x;
  int lane = tid & 31;
  int mt   = tid >> 5;            // wave id = m-tile
  int nt0  = blockIdx.x * 2;      // n-pair

  const unsigned short* wsrc[2] = {
      Wp + (long)(nt0 * 64) * FRAG_USH,
      Wp + (long)((nt0 + 1) * 64) * FRAG_USH };

  auto copy_chunk = [&](int ch, int buf) {
#pragma unroll
    for (int ntp = 0; ntp < 2; ++ntp) {
      const unsigned short* src = wsrc[ntp] + (long)ch * KT_CHUNK * FRAG_USH;
      unsigned int ldsbase =
          (unsigned int)(unsigned long long)&Bs[buf][ntp * KT_CHUNK * FRAG_USH];
#pragma unroll
      for (int i = 0; i < 8; ++i) {            // 128 thr * 16B * 8 = 16 KB
        int off_ush = (i * 128 + tid) * 8;     // 8 ushorts = 16 bytes
        async_g2l_b128(ldsbase + off_ush * 2, src + off_ush);
      }
    }
  };

  copy_chunk(0, 0);
  wait_async0();
  __syncthreads();

  v8f acc0 = {}, acc1 = {};
  for (int ch = 0; ch < 64 / KT_CHUNK; ++ch) {
    if (ch < 64 / KT_CHUNK - 1) copy_chunk(ch + 1, (ch + 1) & 1);
    const unsigned short* b0base = &Bs[ch & 1][0];
    const unsigned short* b1base = &Bs[ch & 1][KT_CHUNK * FRAG_USH];
#pragma unroll 4
    for (int ktl = 0; ktl < KT_CHUNK; ++ktl) {
      int kt = ch * KT_CHUNK + ktl;
      const unsigned short* afrag =
          (kt < 32) ? (Ap + (long)(((t * 4 + mt) * 32) + kt) * FRAG_USH)
                    : (hprev + (long)(mt * 32 + (kt - 32)) * FRAG_USH);
      v16bf a  = load_frag(afrag + lane * 16);
      v16bf b0 = load_frag(b0base + ktl * FRAG_USH + lane * 16);
      v16bf b1 = load_frag(b1base + ktl * FRAG_USH + lane * 16);
      acc0 = __builtin_amdgcn_wmma_f32_16x16x32_bf16(false, a, false, b0,
                                                     (short)0, acc0, false, false);
      acc1 = __builtin_amdgcn_wmma_f32_16x16x32_bf16(false, a, false, b1,
                                                     (short)0, acc1, false, false);
    }
    wait_async0();
    __syncthreads();
  }

  int mhi = (lane >> 4) << 3;
#pragma unroll
  for (int ntp = 0; ntp < 2; ++ntp) {
    int   n    = (nt0 + ntp) * 16 + (lane & 15);
    float bias = bh[n];
    const v8f& acc = ntp ? acc1 : acc0;
#pragma unroll
    for (int v = 0; v < 8; ++v) {
      int bb = mt * 16 + v + mhi;
      gates[(long)bb * LSTM_G + n] = acc[v] + bias;
    }
  }
}

// Pointwise LSTM cell update; writes h_t as packed A-fragments into hs slot.
__global__ void lstm_pointwise_kernel(const float* __restrict__ gates,
                                      float* __restrict__ c,
                                      unsigned short* __restrict__ h_out) {
  int idx = blockIdx.x * blockDim.x + threadIdx.x;   // 64*1024
  int b  = idx >> 10;
  int hh = idx & 1023;
  const float* gr = gates + (long)b * LSTM_G;
  float gi = gr[hh];
  float gf = gr[LSTM_H + hh];
  float gg = gr[2 * LSTM_H + hh];
  float go = gr[3 * LSTM_H + hh];
  gi = 1.0f / (1.0f + __expf(-gi));
  gf = 1.0f / (1.0f + __expf(-gf));
  go = 1.0f / (1.0f + __expf(-go));
  gg = tanhf(gg);
  float cn = gf * c[idx] + gi * gg;
  c[idx] = cn;
  float h = go * tanhf(cn);
  // scatter into A-fragment layout: invert a_kmap
  int mt = b >> 4, m = b & 15;
  int kt = hh >> 5, kl = hh & 31;
  int q    = kl >> 3;
  int lane = m + 16 * (q & 1);
  int j    = (kl & 7) + 8 * (q >> 1);
  h_out[((long)(mt * 32 + kt) * 32 + lane) * 16 + j] = f32_to_bf16_rne(h);
}

// Head GEMM over all steps: logits[b,t,:] = h_t[b,:] @ Wo + bo
__global__ void logits_gemm_kernel(const unsigned short* __restrict__ hs_all,
                                   const unsigned short* __restrict__ Wop,
                                   const float* __restrict__ bo,
                                   float* __restrict__ out) {
  int lane = threadIdx.x & 31;
  int wid  = blockIdx.x * (blockDim.x >> 5) + (threadIdx.x >> 5); // 0..16383
  int nt   = wid & 7;
  int mt   = (wid >> 3) & 3;
  int t    = wid >> 5;
  const unsigned short* abase =
      hs_all + (long)(t + 1) * SLOT_USH + (long)(mt * 32) * FRAG_USH;
  const unsigned short* wbase = Wop + (long)(nt * 32) * FRAG_USH;
  v8f acc = {};
#pragma unroll 4
  for (int kt = 0; kt < 32; ++kt) {
    v16bf a = load_frag(abase + (long)kt * FRAG_USH + lane * 16);
    v16bf b = load_frag(wbase + (long)kt * FRAG_USH + lane * 16);
    acc = __builtin_amdgcn_wmma_f32_16x16x32_bf16(false, a, false, b,
                                                  (short)0, acc, false, false);
  }
  int   n    = nt * 16 + (lane & 15);
  float bias = bo[n];
  int   mhi  = (lane >> 4) << 3;
#pragma unroll
  for (int v = 0; v < 8; ++v) {
    int bg = mt * 16 + v + mhi;
    out[((long)bg * LSTM_S + t) * LSTM_A + n] = acc[v] + bias;
  }
}

// ---------------------------------------------------------------------------
extern "C" void kernel_launch(void* const* d_in, const int* in_sizes, int n_in,
                              void* d_out, int out_size, void* d_ws, size_t ws_size,
                              hipStream_t stream) {
  (void)in_sizes; (void)n_in; (void)out_size; (void)ws_size;
  const float* embed = (const float*)d_in[0];
  const float* Wi    = (const float*)d_in[1];
  const float* Wh    = (const float*)d_in[2];
  const float* bh    = (const float*)d_in[3];
  const float* Wo    = (const float*)d_in[4];
  const float* bo    = (const float*)d_in[5];
  const int*   x     = (const int*)d_in[6];
  // d_in[7..9] (true_actions, true_states, use_forcing) unused in forward.

  // Workspace carve-up (ushort units for packed bf16 buffers)
  unsigned short* Ap  = (unsigned short*)d_ws;                 // 2048*32 frags
  unsigned short* Wp  = Ap  + (long)2048 * 32 * FRAG_USH;      // 256*64 frags
  unsigned short* Wop = Wp  + (long)256 * 64 * FRAG_USH;       // 8*32 frags
  unsigned short* hs  = Wop + (long)8 * 32 * FRAG_USH;         // 513 slots
  float* gates = (float*)(hs + (long)(LSTM_S + 1) * SLOT_USH); // 64*4096 f32
  float* cst   = gates + (long)LSTM_B * LSTM_G;                // 64*1024 f32

  float* out_f      = (float*)d_out;
  float* out_logits = out_f + (long)LSTM_B * LSTM_S;           // after dummy #1
  float* out_dummy2 = out_logits + (long)LSTM_B * LSTM_S * LSTM_A;

  // ---- init: h_0 slot, c state, and the two dummy outputs -> 0
  zero_u32_kernel<<<(SLOT_USH / 2 + 255) / 256, 256, 0, stream>>>(
      (unsigned int*)hs, SLOT_USH / 2);
  zero_u32_kernel<<<(LSTM_B * LSTM_H + 255) / 256, 256, 0, stream>>>(
      (unsigned int*)cst, (long)LSTM_B * LSTM_H);
  zero_u32_kernel<<<(LSTM_B * LSTM_S + 255) / 256, 256, 0, stream>>>(
      (unsigned int*)out_f, (long)LSTM_B * LSTM_S);
  zero_u32_kernel<<<(LSTM_B * LSTM_S + 255) / 256, 256, 0, stream>>>(
      (unsigned int*)out_dummy2, (long)LSTM_B * LSTM_S);

  // ---- one-time operand packing
  gather_pack_A_kernel<<<8192, 256, 0, stream>>>(embed, x, Ap);   // 2M threads
  pack_Wboth_kernel<<<2048, 256, 0, stream>>>(Wi, Wh, Wp);        // 512K threads
  pack_Wo_kernel<<<32, 256, 0, stream>>>(Wo, Wop);                // 8K threads

  // ---- sequential scan: 512 steps x (gates GEMM + pointwise)
  for (int t = 0; t < LSTM_S; ++t) {
    const unsigned short* hprev = hs + (long)t * SLOT_USH;
    unsigned short* hnext = hs + (long)(t + 1) * SLOT_USH;
    lstm_gates_gemm_kernel<<<128, 128, 0, stream>>>(Ap, Wp, hprev, bh, gates, t);
    lstm_pointwise_kernel<<<256, 256, 0, stream>>>(gates, cst, hnext);
  }

  // ---- head GEMM for all (b, t)
  logits_gemm_kernel<<<2048, 256, 0, stream>>>(hs, Wop, bo, out_logits);
}
